// HomoLinear_22196390985760
// MI455X (gfx1250) — compile-verified, matrix operand order checked by reading, not run on hardware
//
#include <hip/hip_runtime.h>
#include <hip/hip_bf16.h>

typedef __attribute__((ext_vector_type(16))) _Float16 v16h;
typedef __attribute__((ext_vector_type(8)))  _Float16 v8h;
typedef __attribute__((ext_vector_type(8)))  float    v8f;
typedef __attribute__((ext_vector_type(4)))  float    v4f;
typedef __attribute__((ext_vector_type(4)))  unsigned int v4u;

#define HID 128
#define WAVES_PER_BLOCK 8
#define THREADS_PER_BLOCK 256

// fragment-pack sizes (halves)
#define W1PACK_HALVES (64 * 32 * 16)   // 32768 (64KB)
#define W2PACK_HALVES (32 * 32 * 16)   // 16384 (32KB)
#define WPACK_HALVES  (W1PACK_HALVES + W2PACK_HALVES)

// ---------------------------------------------------------------------------
// Pre-pass 1: convert node features fp32 -> f16 (one row read per node, once
// per launch; makes every later gather half the bytes and removes all cvt
// VALU from the hot loop).
// ---------------------------------------------------------------------------
__global__ __launch_bounds__(THREADS_PER_BLOCK)
void convert_x_f16(const float* __restrict__ x, _Float16* __restrict__ xh,
                   int nElem8)
{
    int i = blockIdx.x * blockDim.x + threadIdx.x;
    if (i >= nElem8) return;
    const v4f f0 = *(const v4f*)(x + (size_t)i * 8);
    const v4f f1 = *(const v4f*)(x + (size_t)i * 8 + 4);
    v8h h;
#pragma unroll
    for (int k = 0; k < 4; ++k) {
        h[k]     = (_Float16)f0[k];
        h[4 + k] = (_Float16)f1[k];
    }
    *(v8h*)(xh + (size_t)i * 8) = h;
}

// ---------------------------------------------------------------------------
// Pre-pass 2: pack W1/W2 into WMMA B-fragment order (f16) in global scratch.
// One small block; main kernel then fills LDS with a straight b128 copy.
// B element (k,n) of fragment (kt,nt) = W[nt*16+n][kt*32+k_local], where
// k_local follows the ISA 16-bit layout (lanes 0-15: K 0-7/16-23,
// lanes 16-31: K 8-15/24-31).
// ---------------------------------------------------------------------------
__global__ __launch_bounds__(THREADS_PER_BLOCK)
void pack_weights(const float* __restrict__ W1, const float* __restrict__ W2,
                  _Float16* __restrict__ wpack)
{
    const int tid = threadIdx.x;
    for (int idx = tid; idx < 64 * 32; idx += THREADS_PER_BLOCK) {
        const int frag = idx >> 5, l = idx & 31;
        const int kt = frag >> 3, nt = frag & 7;
        const int n = l & 15, lb0 = (l >= 16) ? 8 : 0;
        _Float16* dst = wpack + (size_t)idx * 16;
        const float* srcp = W1 + (size_t)(nt * 16 + n) * 256 + kt * 32;
#pragma unroll
        for (int h = 0; h < 8; ++h) dst[h]     = (_Float16)srcp[lb0 + h];
#pragma unroll
        for (int h = 0; h < 8; ++h) dst[8 + h] = (_Float16)srcp[16 + lb0 + h];
    }
    for (int idx = tid; idx < 32 * 32; idx += THREADS_PER_BLOCK) {
        const int frag = idx >> 5, l = idx & 31;
        const int kt = frag >> 3, nt = frag & 7;
        const int n = l & 15, lb0 = (l >= 16) ? 8 : 0;
        _Float16* dst = wpack + W1PACK_HALVES + (size_t)idx * 16;
        const float* srcp = W2 + (size_t)(nt * 16 + n) * 128 + kt * 32;
#pragma unroll
        for (int h = 0; h < 8; ++h) dst[h]     = (_Float16)srcp[lb0 + h];
#pragma unroll
        for (int h = 0; h < 8; ++h) dst[8 + h] = (_Float16)srcp[16 + lb0 + h];
    }
}

// ---------------------------------------------------------------------------
// Main kernel. PREPACK=true: xh/wpack in global scratch (f16 gathers, LDS
// fill is a vector copy). PREPACK=false: self-contained fp32 fallback.
// ---------------------------------------------------------------------------
template <bool PREPACK>
__global__ __launch_bounds__(THREADS_PER_BLOCK)
void edge_mlp_wmma(const float* __restrict__ x,
                   const _Float16* __restrict__ xh,
                   const long long* __restrict__ src_idx,
                   const long long* __restrict__ dst_idx,
                   const float* __restrict__ W1,
                   const float* __restrict__ b1,
                   const float* __restrict__ W2,
                   const float* __restrict__ b2,
                   const _Float16* __restrict__ wpackg,
                   float* __restrict__ out,
                   int nTiles)
{
    __shared__ _Float16 wpack[WPACK_HALVES] __attribute__((aligned(32)));
    __shared__ _Float16 h1stage[WAVES_PER_BLOCK * 16 * HID] __attribute__((aligned(32)));

    const int tid    = threadIdx.x;
    const int wave   = tid >> 5;
    const int lane   = tid & 31;
    const int laneM  = lane & 15;
    const int hiHalf = lane >> 4;
    const int b0     = hiHalf * 8;

    if (PREPACK) {
        // straight vectorized copy of pre-packed fragments into LDS
        const v4u* src = (const v4u*)wpackg;
        v4u* dst = (v4u*)wpack;
        for (int i = tid; i < WPACK_HALVES / 8; i += THREADS_PER_BLOCK)
            dst[i] = src[i];
    } else {
        for (int idx = tid; idx < 64 * 32; idx += THREADS_PER_BLOCK) {
            const int frag = idx >> 5, l = idx & 31;
            const int kt = frag >> 3, nt = frag & 7;
            const int n = l & 15, lb0 = (l >= 16) ? 8 : 0;
            _Float16* dst = &wpack[idx * 16];
            const float* srcp = W1 + (size_t)(nt * 16 + n) * 256 + kt * 32;
#pragma unroll
            for (int h = 0; h < 8; ++h) dst[h]     = (_Float16)srcp[lb0 + h];
#pragma unroll
            for (int h = 0; h < 8; ++h) dst[8 + h] = (_Float16)srcp[16 + lb0 + h];
        }
        for (int idx = tid; idx < 32 * 32; idx += THREADS_PER_BLOCK) {
            const int frag = idx >> 5, l = idx & 31;
            const int kt = frag >> 3, nt = frag & 7;
            const int n = l & 15, lb0 = (l >= 16) ? 8 : 0;
            _Float16* dst = &wpack[W1PACK_HALVES + idx * 16];
            const float* srcp = W2 + (size_t)(nt * 16 + n) * 128 + kt * 32;
#pragma unroll
            for (int h = 0; h < 8; ++h) dst[h]     = (_Float16)srcp[lb0 + h];
#pragma unroll
            for (int h = 0; h < 8; ++h) dst[8 + h] = (_Float16)srcp[16 + lb0 + h];
        }
    }
    __syncthreads();

    float b1v[8], b2v[8];
#pragma unroll
    for (int nt = 0; nt < 8; ++nt) {
        b1v[nt] = b1[nt * 16 + laneM];
        b2v[nt] = b2[nt * 16 + laneM];
    }

    const v16h* w1v = (const v16h*)wpack;
    const v16h* w2v = (const v16h*)(wpack + W1PACK_HALVES);
    _Float16*   h1  = &h1stage[wave * (16 * HID)];

    for (int tile = blockIdx.x * WAVES_PER_BLOCK + wave; tile < nTiles;
         tile += gridDim.x * WAVES_PER_BLOCK) {

        const long long sI = src_idx[tile * 16 + laneM];
        const long long dI = dst_idx[tile * 16 + laneM];

        // ================= Layer 1: [16x256] x [256x128] =================
        v8f acc1[8];
#pragma unroll
        for (int nt = 0; nt < 8; ++nt) acc1[nt] = {};

#pragma unroll
        for (int kt = 0; kt < 8; ++kt) {
            const int koff = (kt & 3) * 32;
            v16h a;
            if (PREPACK) {
                const _Float16* rowh = xh + (size_t)((kt < 4) ? sI : dI) * HID;
                const v8h r1 = *(const v8h*)(rowh + koff + b0);
                const v8h r2 = *(const v8h*)(rowh + koff + 16 + b0);
                a = __builtin_shufflevector(r1, r2, 0, 1, 2, 3, 4, 5, 6, 7,
                                            8, 9, 10, 11, 12, 13, 14, 15);
            } else {
                const float* row = x + (size_t)((kt < 4) ? sI : dI) * HID;
                const v4f f0 = *(const v4f*)(row + koff + b0);
                const v4f f1 = *(const v4f*)(row + koff + b0 + 4);
                const v4f f2 = *(const v4f*)(row + koff + 16 + b0);
                const v4f f3 = *(const v4f*)(row + koff + 16 + b0 + 4);
#pragma unroll
                for (int h = 0; h < 4; ++h) {
                    a[h]      = (_Float16)f0[h];
                    a[4 + h]  = (_Float16)f1[h];
                    a[8 + h]  = (_Float16)f2[h];
                    a[12 + h] = (_Float16)f3[h];
                }
            }
#pragma unroll
            for (int nt = 0; nt < 8; ++nt) {
                const v16h b = w1v[(kt * 8 + nt) * 32 + lane];
                acc1[nt] = __builtin_amdgcn_wmma_f32_16x16x32_f16(
                    false, a, false, b, (short)0, acc1[nt], false, false);
            }
        }

        // bias + relu, stage h1 to LDS (f16, [m][k]) for layer-2 A reads
#pragma unroll
        for (int nt = 0; nt < 8; ++nt) {
            const float bias = b1v[nt];
#pragma unroll
            for (int r = 0; r < 8; ++r) {
                float v = acc1[nt][r] + bias;
                v = v > 0.0f ? v : 0.0f;
                const int m = r + hiHalf * 8;
                h1[m * HID + nt * 16 + laneM] = (_Float16)v;
            }
        }
        // DS ops in-order within a wave; region is wave-private (no barrier).

        // ================= Layer 2: [16x128] x [128x128] =================
        v8f acc2[8];
#pragma unroll
        for (int nt = 0; nt < 8; ++nt) acc2[nt] = {};

#pragma unroll
        for (int kt = 0; kt < 4; ++kt) {
            const _Float16* rowp = &h1[laneM * HID + kt * 32];
            const v8h r1 = *(const v8h*)(rowp + b0);
            const v8h r2 = *(const v8h*)(rowp + 16 + b0);
            const v16h a = __builtin_shufflevector(r1, r2, 0, 1, 2, 3, 4, 5, 6, 7,
                                                   8, 9, 10, 11, 12, 13, 14, 15);
#pragma unroll
            for (int nt = 0; nt < 8; ++nt) {
                const v16h b = w2v[(kt * 8 + nt) * 32 + lane];
                acc2[nt] = __builtin_amdgcn_wmma_f32_16x16x32_f16(
                    false, a, false, b, (short)0, acc2[nt], false, false);
            }
        }

        // ================= Epilogue: bias + coalesced f32 stores ==========
        float* outp = out + (size_t)tile * 16 * HID;
#pragma unroll
        for (int nt = 0; nt < 8; ++nt) {
            const float bias = b2v[nt];
#pragma unroll
            for (int r = 0; r < 8; ++r) {
                const int m = r + hiHalf * 8;
                outp[m * HID + nt * 16 + laneM] = acc2[nt][r] + bias;
            }
        }
    }
}

extern "C" void kernel_launch(void* const* d_in, const int* in_sizes, int n_in,
                              void* d_out, int out_size, void* d_ws, size_t ws_size,
                              hipStream_t stream) {
    const float*     xp  = (const float*)d_in[0];
    const long long* si  = (const long long*)d_in[1];
    const long long* di  = (const long long*)d_in[2];
    const float*     W1  = (const float*)d_in[3];
    const float*     b1  = (const float*)d_in[4];
    const float*     W2  = (const float*)d_in[5];
    const float*     b2  = (const float*)d_in[6];
    float*           out = (float*)d_out;

    const int nNodeElems = in_sizes[0];          // N_NODES * HID
    const int E          = in_sizes[1];
    const int nTiles     = E / 16;

    int blocks = 1024;
    const int needed = (nTiles + WAVES_PER_BLOCK - 1) / WAVES_PER_BLOCK;
    if (blocks > needed) blocks = needed;

    const size_t xhBytes    = (size_t)nNodeElems * sizeof(_Float16);
    const size_t xhBytesAl  = (xhBytes + 255) & ~(size_t)255;
    const size_t wsNeeded   = xhBytesAl + (size_t)WPACK_HALVES * sizeof(_Float16);

    if (ws_size >= wsNeeded) {
        _Float16* xh    = (_Float16*)d_ws;
        _Float16* wpack = (_Float16*)((char*)d_ws + xhBytesAl);

        const int nElem8 = nNodeElems / 8;
        convert_x_f16<<<(nElem8 + THREADS_PER_BLOCK - 1) / THREADS_PER_BLOCK,
                        THREADS_PER_BLOCK, 0, stream>>>(xp, xh, nElem8);
        pack_weights<<<1, THREADS_PER_BLOCK, 0, stream>>>(W1, W2, wpack);
        edge_mlp_wmma<true><<<blocks, THREADS_PER_BLOCK, 0, stream>>>(
            xp, xh, si, di, W1, b1, W2, b2, wpack, out, nTiles);
    } else {
        edge_mlp_wmma<false><<<blocks, THREADS_PER_BLOCK, 0, stream>>>(
            xp, nullptr, si, di, W1, b1, W2, b2, nullptr, out, nTiles);
    }
}
